// DifferenceAwareAggregator_2250562863814
// MI455X (gfx1250) — matrix operand order repaired and checked by reading, not compiled, not run on hardware
//
#include <hip/hip_runtime.h>
#include <hip/hip_bf16.h>
#include <math.h>
#include <stdint.h>

// ---------------------------------------------------------------------------
// DifferenceAwareAggregator for MI455X (gfx1250, wave32, WMMA bf16, TDM)
//
//  k0: fp32 -> bf16 conversions; weights are stored TRANSPOSED (Wt[n][k]) so
//      LDS weight panels are contiguous (TDM-friendly, no 16-bit scatter).
//  k1: Qall = h_center_bf16 @ Wq + bq              (generic WMMA GEMM, f32 out)
//  k2: fused per-batch-item kernel:
//        combined=[h_n, h_n-h_c] -> @W1+b1 -> LN -> GELU -> hn (LDS, bf16)
//        K = hn@Wk+bk -> scores = Q.K/8 -> mask/softmax
//        V = hn@Wv+bv -> ctx = attn.V   -> ctx bf16 to workspace
//  k3: out = ctx @ Wo + bo                         (generic WMMA GEMM, f32 out)
// ---------------------------------------------------------------------------

typedef __attribute__((ext_vector_type(16))) __bf16 v16bf;
typedef __attribute__((ext_vector_type(8)))  float  v8f;

#define HID     512
#define NNB     32
#define LN_EPS  1e-5f

// workspace layout (bytes); all weight blobs are transposed: Wt[n=0..511][k]
#define OFF_W1   ((size_t)0)                     // 512 x 1024 bf16 (transposed)
#define OFF_WQ   (OFF_W1 + 1024*512*2)           // 512 x 512 bf16 (transposed)
#define OFF_WK   (OFF_WQ + 512*512*2)
#define OFF_WV   (OFF_WK + 512*512*2)
#define OFF_WO   (OFF_WV + 512*512*2)
#define OFF_HC   (OFF_WO + 512*512*2)            // 8192*512 bf16 (row-major)
#define OFF_Q    (OFF_HC + (size_t)8192*512*2)   // 8192*512 f32
#define OFF_CTX  (OFF_Q  + (size_t)8192*512*4)   // 8192*512 bf16

static __device__ inline v8f zero8() {
  v8f z = {0.f,0.f,0.f,0.f,0.f,0.f,0.f,0.f};
  return z;
}

// A fragment, 16x32 bf16 tile (ISA 7.12.2 16-bit A layout).
// lane hi=0 : K = 0..7 then 16..23 ; hi=1 : K = 8..15 then 24..31
static __device__ inline v16bf load_a_frag(const __bf16* base, int lda, int m, int hi) {
  const __bf16* p0 = base + (size_t)m * lda + hi * 8;
  const __bf16* p1 = p0 + 16;
  v16bf a;
#pragma unroll
  for (int i = 0; i < 8; ++i) { a[i] = p0[i]; a[8 + i] = p1[i]; }
  return a;
}

// B fragment, 32x16 bf16 tile from transposed LDS buffer wt[n][32].
// lanes 0-15: K=0..7 then 16..23 at N=lane ; lanes 16-31: K=8..15 then 24..31 at N=lane-16
static __device__ inline v16bf load_b_frag(const __bf16* wt, int nl, int hi) {
  const __bf16* p = wt + (size_t)nl * 32 + hi * 8;
  v16bf b;
#pragma unroll
  for (int i = 0; i < 8; ++i) { b[i] = p[i]; b[8 + i] = p[16 + i]; }
  return b;
}

// ---------------------------------------------------------------------------
// Weight-panel fill: bring rows kc..kc+31 (K) x cols nh*256..+256 (N) of a
// transposed weight matrix Wt[512][Ktot] into LDS wt[256][32] (contiguous).
// Preferred path: Tensor Data Mover (tensor_load_to_lds, 6-arg clang-23 form),
// issued by wave 0, drained with s_wait_tensorcnt. Fallback: b128 copy.
// ---------------------------------------------------------------------------
#if __has_builtin(__builtin_amdgcn_tensor_load_to_lds) && \
    __has_builtin(__builtin_amdgcn_s_wait_tensorcnt)
#define USE_TDM 1
typedef unsigned int u32x4 __attribute__((ext_vector_type(4)));
typedef int          i32x8 __attribute__((ext_vector_type(8)));
typedef int          i32x4 __attribute__((ext_vector_type(4)));

static __device__ inline void tdm_load_panel(__bf16 (*wt)[32], const __bf16* Wt,
                                             int Ktot, int kc, int nh) {
  const __bf16* gsrc = Wt + (size_t)(nh * 256) * Ktot + kc;   // tile start
  const unsigned long long ga = (unsigned long long)(uintptr_t)gsrc;
  // generic LDS pointer: low 32 bits are the LDS byte address (aperture rule)
  const unsigned int la = (unsigned int)(uintptr_t)(&wt[0][0]);

  u32x4 g0;
  g0[0] = 1u;                                        // count=1 (valid user D#)
  g0[1] = la;                                        // lds_addr
  g0[2] = (unsigned int)(ga & 0xffffffffu);          // global_addr[31:0]
  g0[3] = (unsigned int)(ga >> 32) | (2u << 30);     // global_addr[56:32]|type=2

  i32x8 g1;
  g1[0] = (int)(1u << 16);                           // data_size = 1 (2 bytes)
  g1[1] = (int)(((unsigned)Ktot & 0xffffu) << 16);   // tensor_dim0[15:0]
  g1[2] = (int)((((unsigned)Ktot >> 16) & 0xffffu) | (512u << 16)); // dim0 hi | dim1 lo
  g1[3] = (int)(32u << 16);                          // dim1 hi(0) | tile_dim0 = 32
  g1[4] = 256;                                       // tile_dim1 = 256, tile_dim2 = 0
  g1[5] = Ktot;                                      // tensor_dim0_stride[31:0]
  g1[6] = 0;
  g1[7] = 0;

  i32x4 z4 = {0, 0, 0, 0};                           // 2-D tensor: groups 2/3 off
  i32x8 z8 = {0, 0, 0, 0, 0, 0, 0, 0};
  __builtin_amdgcn_tensor_load_to_lds(g0, g1, z4, z4, z8, 0);
}
#else
#define USE_TDM 0
#endif

static __device__ inline void fill_wt(__bf16 (*wt)[32], const __bf16* Wt,
                                      int Ktot, int kc, int nh, int t) {
#if USE_TDM
  if (t < 32) {                       // wave 0 issues the DMA and drains it
    tdm_load_panel(wt, Wt, Ktot, kc, nh);
    __builtin_amdgcn_s_wait_tensorcnt(0);
  }
#else
  const __bf16* src = Wt + (size_t)(nh * 256 + t) * Ktot + kc;  // 32 bf16, 64B
  const float4* sv = (const float4*)src;
  float4* dv = (float4*)&wt[t][0];
#pragma unroll
  for (int i = 0; i < 4; ++i) dv[i] = sv[i];
  if (kc + 32 < Ktot) __builtin_prefetch(src + 32, 0, 1);
#endif
}

// ---------------------------------------------------------------------------
// k0a: elementwise f32 -> bf16 (row-major copy)
// ---------------------------------------------------------------------------
__global__ void cvt_f32_bf16(const float* __restrict__ src,
                             __bf16* __restrict__ dst, int n) {
  int i = blockIdx.x * blockDim.x + threadIdx.x;
  if (i < n) dst[i] = (__bf16)src[i];
}

// k0b: f32 [K x 512] -> bf16 transposed [512 x K]
__global__ void cvt_transpose_bf16(const float* __restrict__ src,
                                   __bf16* __restrict__ dst, int K) {
  int i = blockIdx.x * blockDim.x + threadIdx.x;
  if (i < K * HID) {
    const int k = i / HID, n = i % HID;
    dst[(size_t)n * K + k] = (__bf16)src[i];
  }
}

// ---------------------------------------------------------------------------
// k1/k3: out[M x 512] = A_bf16[M x 512] @ W (Wt transposed [512 x 512]) + bias
// one block = 32 rows, 256 threads = 8 waves, each wave owns 8 16x16 tiles.
// ---------------------------------------------------------------------------
__global__ __launch_bounds__(256) void gemm32_bias(
    const __bf16* __restrict__ A, const __bf16* __restrict__ Wt,
    const float* __restrict__ bias, float* __restrict__ out) {
  __shared__ __bf16 sA[32][32];
  __shared__ __bf16 sW[256][32];

  const int t = threadIdx.x, w = t >> 5, l = t & 31;
  const int hi = l >> 4, nn = l & 15;
  const int row0 = blockIdx.x * 32;

  v8f acc[8];
#pragma unroll
  for (int i = 0; i < 8; ++i) acc[i] = zero8();

  for (int kc = 0; kc < 512; kc += 32) {
    {  // A tile: 32 rows x 32 k ; 4 bf16 (8B) per thread, vectorized
      const int r = t >> 3, k0 = (t & 7) * 4;
      const __bf16* src = A + (size_t)(row0 + r) * HID + kc + k0;
      __bf16 tmp[4];
#pragma unroll
      for (int i = 0; i < 4; ++i) tmp[i] = src[i];
      __builtin_memcpy(&sA[r][k0], tmp, 8);
    }
    __syncthreads();
    for (int nh = 0; nh < 2; ++nh) {
      fill_wt(sW, Wt, 512, kc, nh, t);
      __syncthreads();
#pragma unroll
      for (int mt = 0; mt < 2; ++mt) {
        v16bf aF = load_a_frag(&sA[0][0], 32, mt * 16 + nn, hi);
#pragma unroll
        for (int j = 0; j < 2; ++j) {
          v16bf bF = load_b_frag(&sW[0][0], (w * 2 + j) * 16 + nn, hi);
          const int idx = mt * 4 + nh * 2 + j;
          acc[idx] = __builtin_amdgcn_wmma_f32_16x16x32_bf16(
              false, aF, false, bF, (short)0, acc[idx], false, false);
        }
      }
      __syncthreads();
    }
  }

#pragma unroll
  for (int mt = 0; mt < 2; ++mt)
#pragma unroll
    for (int nh = 0; nh < 2; ++nh)
#pragma unroll
      for (int j = 0; j < 2; ++j) {
        const int col = (nh * 16 + w * 2 + j) * 16 + nn;
        const float bb = bias[col];
        const int idx = mt * 4 + nh * 2 + j;
#pragma unroll
        for (int r = 0; r < 8; ++r) {
          const int m = row0 + mt * 16 + hi * 8 + r;
          out[(size_t)m * HID + col] = acc[idx][r] + bb;
        }
      }
}

// ---------------------------------------------------------------------------
// k2: fused per-batch-item aggregator
// ---------------------------------------------------------------------------
struct __align__(16) SmemFused {
  __bf16 a[32][32];      //  2 KB  A chunk (combined tile, bf16)
  __bf16 wt[256][32];    // 16 KB  weight panel (contiguous, TDM target)
  __bf16 hn[32][512];    // 32 KB  LN+GELU activations
  float  q[512];         //  2 KB  query row
  float  rs[32];         //  row sums (LN)
  float  rq[32];         //  row sums of squares (LN)
  float  sc[8][32];      //  scores / attention weights
  float  ctx[512];       //  context accumulator
};

__global__ __launch_bounds__(256) void fused_agg(
    const float* __restrict__ hC, const float* __restrict__ hN,
    const __bf16* __restrict__ W1t, const float* __restrict__ b1,
    const float* __restrict__ ln_g, const float* __restrict__ ln_b,
    const __bf16* __restrict__ Wkt, const float* __restrict__ bk,
    const __bf16* __restrict__ Wvt, const float* __restrict__ bv,
    const float* __restrict__ Qall, const int* __restrict__ mask,
    __bf16* __restrict__ ctxOut) {
  __shared__ SmemFused s;
  const int b = blockIdx.x;
  const int t = threadIdx.x, w = t >> 5, l = t & 31;
  const int hi = l >> 4, nn = l & 15;

  v8f acc[8];
#pragma unroll
  for (int i = 0; i < 8; ++i) acc[i] = zero8();

  // ---- stage A: preact = [h_n, h_n - h_c] @ W1  (K = 1024) ----
  for (int kc = 0; kc < 1024; kc += 32) {
    {
      const int r = t >> 3, k0 = (t & 7) * 4;
      const int k = kc + k0;                       // 4 consecutive, same half
      __bf16 tmp[4];
      if (k < 512) {
        const float4 v = *(const float4*)(hN + ((size_t)b * NNB + r) * HID + k);
        tmp[0] = (__bf16)v.x; tmp[1] = (__bf16)v.y;
        tmp[2] = (__bf16)v.z; tmp[3] = (__bf16)v.w;
      } else {
        const float4 v = *(const float4*)(hN + ((size_t)b * NNB + r) * HID + (k - 512));
        const float4 c = *(const float4*)(hC + (size_t)b * HID + (k - 512));
        tmp[0] = (__bf16)(v.x - c.x); tmp[1] = (__bf16)(v.y - c.y);
        tmp[2] = (__bf16)(v.z - c.z); tmp[3] = (__bf16)(v.w - c.w);
      }
      __builtin_memcpy(&s.a[r][k0], tmp, 8);
      if (kc + 32 < 512)
        __builtin_prefetch(hN + ((size_t)b * NNB + r) * HID + kc + 32 + k0, 0, 1);
    }
    __syncthreads();
    for (int nh = 0; nh < 2; ++nh) {
      fill_wt(s.wt, W1t, 1024, kc, nh, t);
      __syncthreads();
#pragma unroll
      for (int mt = 0; mt < 2; ++mt) {
        v16bf aF = load_a_frag(&s.a[0][0], 32, mt * 16 + nn, hi);
#pragma unroll
        for (int j = 0; j < 2; ++j) {
          v16bf bF = load_b_frag(&s.wt[0][0], (w * 2 + j) * 16 + nn, hi);
          const int idx = mt * 4 + nh * 2 + j;
          acc[idx] = __builtin_amdgcn_wmma_f32_16x16x32_bf16(
              false, aF, false, bF, (short)0, acc[idx], false, false);
        }
      }
      __syncthreads();
    }
  }

  // ---- bias + per-row mean/var (cross-wave via LDS f32 atomics) ----
  if (t < 32) { s.rs[t] = 0.f; s.rq[t] = 0.f; }
  __syncthreads();
  float lsum[16], lsq[16];
#pragma unroll
  for (int i = 0; i < 16; ++i) { lsum[i] = 0.f; lsq[i] = 0.f; }
#pragma unroll
  for (int mt = 0; mt < 2; ++mt)
#pragma unroll
    for (int nh = 0; nh < 2; ++nh)
#pragma unroll
      for (int j = 0; j < 2; ++j) {
        const int col = (nh * 16 + w * 2 + j) * 16 + nn;
        const float bb = b1[col];
        const int idx = mt * 4 + nh * 2 + j;
#pragma unroll
        for (int r = 0; r < 8; ++r) {
          const float v = acc[idx][r] + bb;
          acc[idx][r] = v;
          lsum[mt * 8 + r] += v;
          lsq[mt * 8 + r]  += v * v;
        }
      }
#pragma unroll
  for (int lm = 0; lm < 16; ++lm) {
    const int m = (lm >> 3) * 16 + hi * 8 + (lm & 7);
    atomicAdd(&s.rs[m], lsum[lm]);
    atomicAdd(&s.rq[m], lsq[lm]);
  }
  __syncthreads();

  // ---- LN + exact GELU -> hn (bf16, resident in LDS) ----
#pragma unroll
  for (int mt = 0; mt < 2; ++mt)
#pragma unroll
    for (int nh = 0; nh < 2; ++nh)
#pragma unroll
      for (int j = 0; j < 2; ++j) {
        const int col = (nh * 16 + w * 2 + j) * 16 + nn;
        const float g = ln_g[col], be = ln_b[col];
        const int idx = mt * 4 + nh * 2 + j;
#pragma unroll
        for (int r = 0; r < 8; ++r) {
          const int m = mt * 16 + hi * 8 + r;
          const float mu  = s.rs[m] * (1.f / 512.f);
          const float var = s.rq[m] * (1.f / 512.f) - mu * mu;
          const float x = (acc[idx][r] - mu) * rsqrtf(var + LN_EPS) * g + be;
          const float ge = 0.5f * x * (1.f + erff(x * 0.70710678118f));
          s.hn[m][col] = (__bf16)ge;
        }
      }

  // ---- load Q row, zero score accumulators ----
  for (int i = t; i < 512; i += 256) s.q[i] = Qall[(size_t)b * HID + i];
  s.sc[t >> 5][t & 31] = 0.f;   // 256 entries
  __syncthreads();

  // ---- stage B1: K = hn @ Wk ; scores[h][n] = (K . Q_h) / 8 ----
#pragma unroll
  for (int i = 0; i < 8; ++i) acc[i] = zero8();
  for (int kc = 0; kc < 512; kc += 32) {
    for (int nh = 0; nh < 2; ++nh) {
      fill_wt(s.wt, Wkt, 512, kc, nh, t);
      __syncthreads();
#pragma unroll
      for (int mt = 0; mt < 2; ++mt) {
        v16bf aF = load_a_frag(&s.hn[0][0] + kc, 512, mt * 16 + nn, hi);
#pragma unroll
        for (int j = 0; j < 2; ++j) {
          v16bf bF = load_b_frag(&s.wt[0][0], (w * 2 + j) * 16 + nn, hi);
          const int idx = mt * 4 + nh * 2 + j;
          acc[idx] = __builtin_amdgcn_wmma_f32_16x16x32_bf16(
              false, aF, false, bF, (short)0, acc[idx], false, false);
        }
      }
      __syncthreads();
    }
  }
#pragma unroll
  for (int mt = 0; mt < 2; ++mt)
#pragma unroll
    for (int nh = 0; nh < 2; ++nh)
#pragma unroll
      for (int j = 0; j < 2; ++j) {
        const int col = (nh * 16 + w * 2 + j) * 16 + nn;
        const float bb = bk[col];
        const float qv = s.q[col];
        const int idx = mt * 4 + nh * 2 + j;
#pragma unroll
        for (int r = 0; r < 8; ++r) {
          const int m = mt * 16 + hi * 8 + r;
          atomicAdd(&s.sc[col >> 6][m], (acc[idx][r] + bb) * qv);
        }
      }
  __syncthreads();

  // ---- mask + softmax over neighbors (8 heads, one thread each) ----
  if (t < 8) {
    const int h = t;
    float tmp[32];
    float mx = -3.402823e38f;
#pragma unroll
    for (int n = 0; n < 32; ++n) {
      float sv = s.sc[h][n] * 0.125f;                 // 1/sqrt(64)
      if (mask[(size_t)b * NNB + n] == 0) sv = -3.402823e38f;
      tmp[n] = sv;
      mx = fmaxf(mx, sv);
    }
    float sum = 0.f;
#pragma unroll
    for (int n = 0; n < 32; ++n) { tmp[n] = __expf(tmp[n] - mx); sum += tmp[n]; }
    const float inv = 1.f / sum;
#pragma unroll
    for (int n = 0; n < 32; ++n) s.sc[h][n] = tmp[n] * inv;
  }
  for (int i = t; i < 512; i += 256) s.ctx[i] = 0.f;
  __syncthreads();

  // ---- stage B2: V = hn @ Wv ; ctx[c] = sum_n attn[c>>6][n] * V[n][c] ----
#pragma unroll
  for (int i = 0; i < 8; ++i) acc[i] = zero8();
  for (int kc = 0; kc < 512; kc += 32) {
    for (int nh = 0; nh < 2; ++nh) {
      fill_wt(s.wt, Wvt, 512, kc, nh, t);
      __syncthreads();
#pragma unroll
      for (int mt = 0; mt < 2; ++mt) {
        v16bf aF = load_a_frag(&s.hn[0][0] + kc, 512, mt * 16 + nn, hi);
#pragma unroll
        for (int j = 0; j < 2; ++j) {
          v16bf bF = load_b_frag(&s.wt[0][0], (w * 2 + j) * 16 + nn, hi);
          const int idx = mt * 4 + nh * 2 + j;
          acc[idx] = __builtin_amdgcn_wmma_f32_16x16x32_bf16(
              false, aF, false, bF, (short)0, acc[idx], false, false);
        }
      }
      __syncthreads();
    }
  }
#pragma unroll
  for (int mt = 0; mt < 2; ++mt)
#pragma unroll
    for (int nh = 0; nh < 2; ++nh)
#pragma unroll
      for (int j = 0; j < 2; ++j) {
        const int col = (nh * 16 + w * 2 + j) * 16 + nn;
        const float bb = bv[col];
        const int idx = mt * 4 + nh * 2 + j;
#pragma unroll
        for (int r = 0; r < 8; ++r) {
          const int m = mt * 16 + hi * 8 + r;
          atomicAdd(&s.ctx[col], (acc[idx][r] + bb) * s.sc[col >> 6][m]);
        }
      }
  __syncthreads();

  for (int i = t; i < 512; i += 256)
    ctxOut[(size_t)b * HID + i] = (__bf16)s.ctx[i];
}

// ---------------------------------------------------------------------------
extern "C" void kernel_launch(void* const* d_in, const int* in_sizes, int n_in,
                              void* d_out, int out_size, void* d_ws, size_t ws_size,
                              hipStream_t stream) {
  const float* h_center    = (const float*)d_in[0];
  const float* h_neighbors = (const float*)d_in[1];
  const float* W1   = (const float*)d_in[2];
  const float* b1   = (const float*)d_in[3];
  const float* ln_g = (const float*)d_in[4];
  const float* ln_b = (const float*)d_in[5];
  const float* Wq   = (const float*)d_in[6];
  const float* bq   = (const float*)d_in[7];
  const float* Wk   = (const float*)d_in[8];
  const float* bk   = (const float*)d_in[9];
  const float* Wv   = (const float*)d_in[10];
  const float* bv   = (const float*)d_in[11];
  const float* Wo   = (const float*)d_in[12];
  const float* bo   = (const float*)d_in[13];
  const int*   mask = (const int*)d_in[14];

  char* ws = (char*)d_ws;
  __bf16* W1t  = (__bf16*)(ws + OFF_W1);
  __bf16* Wqt  = (__bf16*)(ws + OFF_WQ);
  __bf16* Wkt  = (__bf16*)(ws + OFF_WK);
  __bf16* Wvt  = (__bf16*)(ws + OFF_WV);
  __bf16* Wot  = (__bf16*)(ws + OFF_WO);
  __bf16* hcbf = (__bf16*)(ws + OFF_HC);
  float*  Qall = (float*)(ws + OFF_Q);
  __bf16* ctxb = (__bf16*)(ws + OFF_CTX);

  const int B = 8192;

  // k0: conversions (weights transposed for contiguous LDS panels)
  cvt_transpose_bf16<<<(1024 * 512 + 255) / 256, 256, 0, stream>>>(W1, W1t, 1024);
  cvt_transpose_bf16<<<(512 * 512 + 255) / 256, 256, 0, stream>>>(Wq, Wqt, 512);
  cvt_transpose_bf16<<<(512 * 512 + 255) / 256, 256, 0, stream>>>(Wk, Wkt, 512);
  cvt_transpose_bf16<<<(512 * 512 + 255) / 256, 256, 0, stream>>>(Wv, Wvt, 512);
  cvt_transpose_bf16<<<(512 * 512 + 255) / 256, 256, 0, stream>>>(Wo, Wot, 512);
  cvt_f32_bf16<<<(B * 512 + 255) / 256, 256, 0, stream>>>(h_center, hcbf, B * 512);

  // k1: Q = h_center @ Wq + bq
  gemm32_bias<<<B / 32, 256, 0, stream>>>(hcbf, Wqt, bq, Qall);

  // k2: fused diff-projection + LN + GELU + single-query MHA
  fused_agg<<<B, 256, 0, stream>>>(h_center, h_neighbors, W1t, b1, ln_g, ln_b,
                                   Wkt, bk, Wvt, bv, Qall, mask, ctxb);

  // k3: out = ctx @ Wo + bo
  gemm32_bias<<<B / 32, 256, 0, stream>>>(ctxb, Wot, bo, (float*)d_out);
}
